// DeepHybridTRM_34591666602459
// MI455X (gfx1250) — matrix operand order repaired and checked by reading, not compiled, not run on hardware
//
#include <hip/hip_runtime.h>
#include <hip/hip_bf16.h>

typedef __attribute__((ext_vector_type(16))) _Float16 v16h;
typedef __attribute__((ext_vector_type(8)))  _Float16 v8h;
typedef __attribute__((ext_vector_type(8)))  float    v8f;

// Model dims (from reference)
static constexpr int kNP    = 22;
static constexpr int kSD    = 6;
static constexpr int kM2V   = 200;
static constexpr int kDA    = 40;
static constexpr int kDH    = 64;
static constexpr int kFF    = 96;
static constexpr int kSTEPS = 16;
static constexpr int kDHD   = 10;   // DA / NH
static constexpr int kXW    = kNP * kSD + kM2V;  // 332

// Blocking: one wave (32 threads) per block, 8 samples per block
static constexpr int kG    = 8;
static constexpr int kROWS = kG * kNP;  // 176 = 11 WMMA M-tiles

// ---------------------------------------------------------------------------
// Weight pack kernel: f32 (fout,fin) row-major  ->  f16 WMMA B-operand tiles.
// Tile (kt,nt): 32 lanes x 16 halfs. lane holds column n = nt*16 + (lane&15);
// lanes 0-15 hold K = kt*32 + {0..7,16..23}, lanes 16-31 hold K = kt*32 + {8..15,24..31}.
// Out-of-range k/n are packed as 0 -> A-side K-overreads contribute 0.
// ---------------------------------------------------------------------------
__global__ __launch_bounds__(32)
void pack_weight(const float* __restrict__ W, int fout, int fin, int rowOff,
                 int NT, _Float16* __restrict__ dst) {
  int tile = blockIdx.x;               // kt*NT + nt
  int nt = tile % NT;
  int lane = threadIdx.x;
  int n  = nt * 16 + (lane & 15);
  int kb = (tile / NT) * 32 + ((lane & 16) ? 8 : 0);
#pragma unroll
  for (int e = 0; e < 16; ++e) {
    int k = kb + (e < 8 ? e : e + 8);
    float v = (k < fin && n < fout) ? W[(size_t)(n + rowOff) * fin + k] : 0.0f;
    dst[(size_t)tile * 512 + lane * 16 + e] = (_Float16)v;
  }
}

// ---------------------------------------------------------------------------
// Parameter bundle
// ---------------------------------------------------------------------------
struct Params {
  const float* x;
  long long Btotal;
  const float *tok_b, *tok_g, *tok_be;
  const float *m2v_b, *m2v_g, *m2v_be;
  const float *sa1_bqkv, *sa1_bo, *sa1n_g, *sa1n_b;
  const float *sa2_bqkv, *sa2_bo, *sa2n_g, *sa2n_b;
  const float *ca_bqkv, *ca_bo, *can_g, *can_b;
  const float *sa1_f1b, *sa1_f2b, *sa1fn_g, *sa1fn_b;
  const float *sa2_f1b, *sa2_f2b, *sa2fn_g, *sa2fn_b;
  const float *pool_b, *pool_g, *pool_be;
  const float *zu1b, *zu2b, *zun_g, *zun_b;
  const float *yu1b, *yu2b, *yun_g, *yun_b;
  const float *head_W, *head_b;
  const _Float16 *w_tok, *w_m2v;
  const _Float16 *w_sa1_qkv, *w_sa1_o, *w_sa1_f1, *w_sa1_f2;
  const _Float16 *w_sa2_qkv, *w_sa2_o, *w_sa2_f1, *w_sa2_f2;
  const _Float16 *w_ca_v, *w_ca_o;
  const _Float16 *w_pool, *w_zu1, *w_zu2, *w_yu1, *w_yu2;
  float* out;
};

// ---------------------------------------------------------------------------
// Unconditional A-operand gather: two contiguous 16B LDS reads per lane.
// 16-bit A 16x32 layout: lane row m = m0 + (lane&15);
// element e -> K = k0 + (lane&16 ? 8 : 0) + (e<8 ? e : e+8)
//  => halves [kb, kb+7] and [kb+16, kb+23], each contiguous & 16B aligned
//     (requires stride % 8 == 0, which all our buffers satisfy).
// ---------------------------------------------------------------------------
__device__ __forceinline__ v16h load_A(const _Float16* src, int stride, int m0,
                                       int k0) {
  int lane = threadIdx.x;
  int m  = m0 + (lane & 15);
  int kb = k0 + ((lane & 16) ? 8 : 0);
  const _Float16* p = src + m * stride + kb;
  v8h lo = *(const v8h*)(p);
  v8h hi = *(const v8h*)(p + 16);
  return __builtin_shufflevector(lo, hi, 0, 1, 2, 3, 4, 5, 6, 7,
                                 8, 9, 10, 11, 12, 13, 14, 15);
}

// D = A(LDS, MT*16 rows) * Wpacked + bias -> LDS f16 (strided).
// All-template tile counts: bias preloaded to registers, N-guard folds for
// exact tiles, and every tile writes its full 16 rows (no row guard; all
// destination regions have 16-row capacity and garbage rows are finite).
template <int KT, int NT, int MT, int NOUT>
__device__ __forceinline__ void gemm(const _Float16* __restrict__ src, int sstride,
                                     const _Float16* __restrict__ wpk,
                                     _Float16* __restrict__ dst, int dstride,
                                     const float* __restrict__ bias) {
  int lane = threadIdx.x;
  float bv[NT];
#pragma unroll
  for (int nt = 0; nt < NT; ++nt) {
    int n = nt * 16 + (lane & 15);
    bv[nt] = (n < NOUT) ? bias[n] : 0.0f;
  }
#pragma unroll 1
  for (int mt = 0; mt < MT; ++mt) {
    v16h a[KT];
#pragma unroll
    for (int kt = 0; kt < KT; ++kt) a[kt] = load_A(src, sstride, mt * 16, kt * 32);
#pragma unroll
    for (int nt = 0; nt < NT; ++nt) {
      v8f acc = {};
#pragma unroll
      for (int kt = 0; kt < KT; ++kt) {
        v16h b = *(const v16h*)(wpk + (size_t)(kt * NT + nt) * 512 + lane * 16);
        acc = __builtin_amdgcn_wmma_f32_16x16x32_f16(
            false, a[kt], false, b, (short)0, acc, false, false);
      }
      int n = nt * 16 + (lane & 15);
      if (n < NOUT) {
        _Float16* dp = dst + (mt * 16 + ((lane & 16) ? 8 : 0)) * dstride + n;
#pragma unroll
        for (int r = 0; r < 8; ++r) dp[r * dstride] = (_Float16)(acc[r] + bv[nt]);
      }
    }
  }
}

__device__ __forceinline__ float gelu_f(float x) {
  return 0.5f * x * (1.0f + erff(x * 0.70710678118654752f));
}

__device__ void ln_rows(_Float16* buf, int stride, int rows, int F,
                        const float* g, const float* b, bool postGelu) {
  for (int r = threadIdx.x; r < rows; r += 32) {
    _Float16* p = buf + r * stride;
    float m = 0.f;
    for (int f = 0; f < F; ++f) m += (float)p[f];
    m /= F;
    float v = 0.f;
    for (int f = 0; f < F; ++f) { float d = (float)p[f] - m; v += d * d; }
    float inv = rsqrtf(v / F + 1e-5f);
    for (int f = 0; f < F; ++f) {
      float y = ((float)p[f] - m) * inv * g[f] + b[f];
      if (postGelu) y = gelu_f(y);
      p[f] = (_Float16)y;
    }
  }
}

// buf = LN(buf + add); add row = r / rowDiv (rowDiv=kNP broadcasts per-sample)
__device__ void ln_res_rows(_Float16* buf, int stride, const _Float16* add,
                            int astride, int rows, int rowDiv, int F,
                            const float* g, const float* b) {
  for (int r = threadIdx.x; r < rows; r += 32) {
    _Float16* p = buf + r * stride;
    const _Float16* q = add + (r / rowDiv) * astride;
    float m = 0.f;
    for (int f = 0; f < F; ++f) m += (float)p[f] + (float)q[f];
    m /= F;
    float v = 0.f;
    for (int f = 0; f < F; ++f) {
      float d = (float)p[f] + (float)q[f] - m;
      v += d * d;
    }
    float inv = rsqrtf(v / F + 1e-5f);
    for (int f = 0; f < F; ++f) {
      float y = ((float)p[f] + (float)q[f] - m) * inv * g[f] + b[f];
      p[f] = (_Float16)y;
    }
  }
}

__device__ void gelu_all(_Float16* p, int count) {
  for (int i = threadIdx.x; i < count; i += 32) p[i] = (_Float16)gelu_f((float)p[i]);
}

// Per-(sample,head) attention over the LDS qkv buffer [176][120].
// Lane = s*4 + h (32 tasks exactly). Output overwrites the Q slot.
__device__ void attention(_Float16* qkv) {
  int lane = threadIdx.x;
  int s = lane >> 2, h = lane & 3;
  _Float16* base = qkv + s * kNP * 120 + h * kDHD;
  const float scale = 0.31622776601683794f;  // 1/sqrt(10)
  for (int t = 0; t < kNP; ++t) {
    float q[kDHD];
#pragma unroll
    for (int d = 0; d < kDHD; ++d) q[d] = (float)base[t * 120 + d];
    float sc[kNP];
    float mx = -1e30f;
#pragma unroll
    for (int u = 0; u < kNP; ++u) {
      const _Float16* kr = base + u * 120 + kDA;
      float dot = 0.f;
#pragma unroll
      for (int d = 0; d < kDHD; ++d) dot += q[d] * (float)kr[d];
      dot *= scale;
      sc[u] = dot;
      mx = fmaxf(mx, dot);
    }
    float den = 0.f;
#pragma unroll
    for (int u = 0; u < kNP; ++u) { float e = expf(sc[u] - mx); sc[u] = e; den += e; }
    float o[kDHD];
#pragma unroll
    for (int d = 0; d < kDHD; ++d) o[d] = 0.f;
#pragma unroll
    for (int u = 0; u < kNP; ++u) {
      const _Float16* vr = base + u * 120 + 2 * kDA;
      float w = sc[u];
#pragma unroll
      for (int d = 0; d < kDHD; ++d) o[d] += w * (float)vr[d];
    }
    float rden = 1.f / den;
#pragma unroll
    for (int d = 0; d < kDHD; ++d) base[t * 120 + d] = (_Float16)(o[d] * rden);
  }
}

// ---------------------------------------------------------------------------
// Fused forward kernel: one wave per 8 samples.
// ---------------------------------------------------------------------------
static constexpr int kMGStride = 32;                 // kSD padded to 32
static constexpr int kMGHalfs  = kROWS * kMGStride;  // 5632
static constexpr int kM2VOff   = kMGHalfs;           // m2v staged after mg
static constexpr int kM2VHalfs = 16 * kM2V + 32;     // 16 rows + slack = 3232

__global__ __launch_bounds__(32)
void deep_hybrid_trm_kernel(Params P) {
  __shared__ __attribute__((aligned(16))) _Float16 sTok[kROWS * kDA + 32];
  __shared__ __attribute__((aligned(16))) _Float16 sBuf[kROWS * 120 + 32];
  __shared__ __attribute__((aligned(16))) _Float16 sCtx[16 * kDA + 32];

  int lane = threadIdx.x;
  long long s0 = (long long)blockIdx.x * kG;

  // ---- zero staging / pad regions (NaN-safe K-overreads) ----
  {
    uint32_t* z = (uint32_t*)sBuf;
    for (int i = lane; i < (kMGHalfs + kM2VHalfs) / 2; i += 32) z[i] = 0u;
    uint32_t* zc = (uint32_t*)sCtx;
    for (int i = lane; i < (16 * kDA + 32) / 2; i += 32) zc[i] = 0u;
    uint32_t* zt = (uint32_t*)(sTok + kROWS * kDA);
    if (lane < 16) zt[lane] = 0u;
  }
  __syncthreads();

  // ---- stage input: mg -> [176][32] (padded), m2v -> [16][200] ----
  _Float16* sMG  = sBuf;
  _Float16* sM2V = sBuf + kM2VOff;
  for (int idx = lane; idx < kG * kXW; idx += 32) {
    int s = idx / kXW, j = idx % kXW;
    float v = (s0 + s < P.Btotal) ? P.x[(size_t)(s0 + s) * kXW + j] : 0.0f;
    if (j < kNP * kSD) sMG[(s * kNP + j / kSD) * kMGStride + (j % kSD)] = (_Float16)v;
    else               sM2V[s * kM2V + (j - kNP * kSD)] = (_Float16)v;
  }
  __syncthreads();

  // ---- token embed: gelu(ln(mg @ tok_W.T + b)) ----
  gemm<1, 3, 11, kDA>(sMG, kMGStride, P.w_tok, sTok, kDA, P.tok_b);
  ln_rows(sTok, kDA, kROWS, kDA, P.tok_g, P.tok_be, true);

  // ---- ctx embed: gelu(ln(m2v @ m2v_W.T + b)) ----
  gemm<7, 3, 1, kDA>(sM2V, kM2V, P.w_m2v, sCtx, kDA, P.m2v_b);
  ln_rows(sCtx, kDA, kG, kDA, P.m2v_g, P.m2v_be, true);
  __syncthreads();

  // ---- two self-attention + FF blocks ----
  for (int L = 0; L < 2; ++L) {
    const _Float16* wq  = L ? P.w_sa2_qkv : P.w_sa1_qkv;
    const _Float16* wo  = L ? P.w_sa2_o   : P.w_sa1_o;
    const _Float16* wf1 = L ? P.w_sa2_f1  : P.w_sa1_f1;
    const _Float16* wf2 = L ? P.w_sa2_f2  : P.w_sa1_f2;
    const float* bqkv = L ? P.sa2_bqkv : P.sa1_bqkv;
    const float* bo   = L ? P.sa2_bo   : P.sa1_bo;
    const float* ng   = L ? P.sa2n_g   : P.sa1n_g;
    const float* nb   = L ? P.sa2n_b   : P.sa1n_b;
    const float* f1b  = L ? P.sa2_f1b  : P.sa1_f1b;
    const float* f2b  = L ? P.sa2_f2b  : P.sa1_f2b;
    const float* fng  = L ? P.sa2fn_g  : P.sa1fn_g;
    const float* fnb  = L ? P.sa2fn_b  : P.sa1fn_b;

    // qkv: 176x40 -> 176x120 (fills all of sBuf with finite f16)
    gemm<2, 8, 11, 120>(sTok, kDA, wq, sBuf, 120, bqkv);
    __syncthreads();
    attention(sBuf);                       // out overwrites Q cols [0,40)
    __syncthreads();
    // Wo: reads cols [0,64) (attn out + k, B zero-padded), writes cols [80,120)
    gemm<2, 3, 11, kDA>(sBuf, 120, wo, sBuf + 2 * kDA, 120, bo);
    ln_res_rows(sTok, kDA, sBuf + 2 * kDA, 120, kROWS, 1, kDA, ng, nb);
    __syncthreads();
    // FF: 40 -> 80 (gelu) -> 40
    gemm<2, 5, 11, 80>(sTok, kDA, wf1, sBuf, 80, f1b);
    gelu_all(sBuf, kROWS * 80);
    __syncthreads();
    gemm<3, 3, 11, kDA>(sBuf, 80, wf2, sBuf + kROWS * 80, kDA, f2b);
    ln_res_rows(sTok, kDA, sBuf + kROWS * 80, kDA, kROWS, 1, kDA, fng, fnb);
    __syncthreads();
  }

  // ---- cross-attention: Sk==1 => softmax==1 => out = (ctx@Wv+bv)@Wo+bo ----
  gemm<2, 3, 1, kDA>(sCtx, kDA, P.w_ca_v, sBuf, kDA, P.ca_bqkv + 2 * kDA);
  __syncthreads();
  gemm<2, 3, 1, kDA>(sBuf, kDA, P.w_ca_o, sBuf + 1024, kDA, P.ca_bo);
  ln_res_rows(sTok, kDA, sBuf + 1024, kDA, kROWS, kNP, kDA, P.can_g, P.can_b);
  __syncthreads();

  // ---- pool + projection ----
  _Float16* sPool = sBuf;          // 16x40 region (rows >=8 hold finite junk)
  _Float16* sXp   = sBuf + 1024;   // 16x64
  for (int idx = lane; idx < kG * kDA; idx += 32) {
    int s = idx / kDA, f = idx % kDA;
    float acc = 0.f;
    for (int t = 0; t < kNP; ++t) acc += (float)sTok[(s * kNP + t) * kDA + f];
    sPool[idx] = (_Float16)(acc * (1.0f / kNP));
  }
  __syncthreads();
  gemm<2, 4, 1, kDH>(sPool, kDA, P.w_pool, sXp, kDH, P.pool_b);
  ln_rows(sXp, kDH, kG, kDH, P.pool_g, P.pool_be, true);
  __syncthreads();

  // ---- recurrent refinement (16 steps) ----
  _Float16* sH = sBuf + 2048;                 // 16x192  [2048, 5120)
  _Float16* sU = sBuf + 5120;                 // 16x96   [5120, 6656)
  _Float16* sV = sBuf + 6656;                 // 16x64   [6656, 7680)
  float* sZ = (float*)(sBuf + 7680);          // 8x64 f32
  float* sY = (float*)(sBuf + 8704);          // 8x64 f32
  for (int i = lane; i < kG * kDH; i += 32) { sZ[i] = 0.f; sY[i] = 0.f; }
  __syncthreads();

  for (int step = 0; step < kSTEPS; ++step) {
    // h = [xp, y, z]
    for (int i = lane; i < kG * 3 * kDH; i += 32) {
      int s = i / (3 * kDH), j = i % (3 * kDH);
      float v = (j < kDH)     ? (float)sXp[s * kDH + j]
              : (j < 2 * kDH) ? sY[s * kDH + j - kDH]
                              : sZ[s * kDH + j - 2 * kDH];
      sH[i] = (_Float16)v;
    }
    __syncthreads();
    gemm<6, 6, 1, kFF>(sH, 3 * kDH, P.w_zu1, sU, kFF, P.zu1b);
    gelu_all(sU, kG * kFF);
    __syncthreads();
    gemm<3, 4, 1, kDH>(sU, kFF, P.w_zu2, sV, kDH, P.zu2b);
    ln_rows(sV, kDH, kG, kDH, P.zun_g, P.zun_b, false);
    for (int i = lane; i < kG * kDH; i += 32) sZ[i] += (float)sV[i];
    __syncthreads();
    // h2 = [y, z]
    for (int i = lane; i < kG * 2 * kDH; i += 32) {
      int s = i / (2 * kDH), j = i % (2 * kDH);
      float v = (j < kDH) ? sY[s * kDH + j] : sZ[s * kDH + j - kDH];
      sH[i] = (_Float16)v;
    }
    __syncthreads();
    gemm<4, 6, 1, kFF>(sH, 2 * kDH, P.w_yu1, sU, kFF, P.yu1b);
    gelu_all(sU, kG * kFF);
    __syncthreads();
    gemm<3, 4, 1, kDH>(sU, kFF, P.w_yu2, sV, kDH, P.yu2b);
    ln_rows(sV, kDH, kG, kDH, P.yun_g, P.yun_b, false);
    for (int i = lane; i < kG * kDH; i += 32) sY[i] += (float)sV[i];
    __syncthreads();
  }

  // ---- head: y @ head_W.T + b ----
  if (lane < kG && s0 + lane < P.Btotal) {
    float acc = P.head_b[0];
    for (int f = 0; f < kDH; ++f) acc += sY[lane * kDH + f] * P.head_W[f];
    P.out[s0 + lane] = acc;
  }
}

// ---------------------------------------------------------------------------
// Host launch
// ---------------------------------------------------------------------------
extern "C" void kernel_launch(void* const* d_in, const int* in_sizes, int n_in,
                              void* d_out, int out_size, void* d_ws, size_t ws_size,
                              hipStream_t stream) {
  (void)n_in; (void)out_size; (void)ws_size;
  auto f = [&](int i) { return (const float*)d_in[i]; };

  _Float16* wbase = (_Float16*)d_ws;
  size_t off = 0;
  auto pack = [&](int idx, int fout, int fin, int rowOff) -> const _Float16* {
    int KT = (fin + 31) / 32, NT = (fout + 15) / 16;
    _Float16* dst = wbase + off;
    hipLaunchKernelGGL(pack_weight, dim3(KT * NT), dim3(32), 0, stream,
                       (const float*)d_in[idx], fout, fin, rowOff, NT, dst);
    off += (size_t)KT * NT * 512;
    return dst;
  };

  Params P;
  P.x = f(0);
  P.Btotal = (long long)(in_sizes[0] / kXW);

  P.w_tok = pack(1, kDA, kSD, 0);
  P.tok_b = f(2); P.tok_g = f(3); P.tok_be = f(4);
  P.w_m2v = pack(5, kDA, kM2V, 0);
  P.m2v_b = f(6); P.m2v_g = f(7); P.m2v_be = f(8);

  P.w_sa1_qkv = pack(9, 3 * kDA, kDA, 0);
  P.sa1_bqkv = f(10);
  P.w_sa1_o = pack(11, kDA, kDA, 0);
  P.sa1_bo = f(12); P.sa1n_g = f(13); P.sa1n_b = f(14);

  P.w_sa2_qkv = pack(15, 3 * kDA, kDA, 0);
  P.sa2_bqkv = f(16);
  P.w_sa2_o = pack(17, kDA, kDA, 0);
  P.sa2_bo = f(18); P.sa2n_g = f(19); P.sa2n_b = f(20);

  P.w_ca_v = pack(21, kDA, kDA, 2 * kDA);  // rows 80..119 of ca_Wqkv (== Wv)
  P.ca_bqkv = f(22);
  P.w_ca_o = pack(23, kDA, kDA, 0);
  P.ca_bo = f(24); P.can_g = f(25); P.can_b = f(26);

  P.w_sa1_f1 = pack(27, 2 * kDA, kDA, 0);
  P.sa1_f1b = f(28);
  P.w_sa1_f2 = pack(29, kDA, 2 * kDA, 0);
  P.sa1_f2b = f(30); P.sa1fn_g = f(31); P.sa1fn_b = f(32);

  P.w_sa2_f1 = pack(33, 2 * kDA, kDA, 0);
  P.sa2_f1b = f(34);
  P.w_sa2_f2 = pack(35, kDA, 2 * kDA, 0);
  P.sa2_f2b = f(36); P.sa2fn_g = f(37); P.sa2fn_b = f(38);

  P.w_pool = pack(39, kDH, kDA, 0);
  P.pool_b = f(40); P.pool_g = f(41); P.pool_be = f(42);

  P.w_zu1 = pack(43, kFF, 3 * kDH, 0);
  P.zu1b = f(44);
  P.w_zu2 = pack(45, kDH, kFF, 0);
  P.zu2b = f(46); P.zun_g = f(47); P.zun_b = f(48);

  P.w_yu1 = pack(49, kFF, 2 * kDH, 0);
  P.yu1b = f(50);
  P.w_yu2 = pack(51, kDH, kFF, 0);
  P.yu2b = f(52); P.yun_g = f(53); P.yun_b = f(54);

  P.head_W = f(55); P.head_b = f(56);
  P.out = (float*)d_out;

  long long nBlocks = (P.Btotal + kG - 1) / kG;
  hipLaunchKernelGGL(deep_hybrid_trm_kernel, dim3((unsigned)nBlocks), dim3(32),
                     0, stream, P);
}
